// MLPF_73418170958086
// MI455X (gfx1250) — matrix operand order repaired and checked by reading, not compiled, not run on hardware
//
#include <hip/hip_runtime.h>
#include <hip/hip_bf16.h>
#include <math.h>

// ---------------- types ----------------
typedef __bf16 bf16_t;
typedef __attribute__((ext_vector_type(16))) __bf16 v16bf;
typedef __attribute__((ext_vector_type(8)))  float  v8f;
typedef __attribute__((ext_vector_type(4)))  int    v4i;

#define NQ        65536        // total nodes
#define NGRAPH    64
#define GSIZE     1024
#define MT        128          // GEMM M tile
#define NT        64           // GEMM N tile
#define KC        64           // K staged per barrier (2 WMMA K-steps)

// ---- CDNA5 async global->LDS path (ASYNCcnt) ----
#if defined(__has_builtin)
#  if __has_builtin(__builtin_amdgcn_global_load_async_to_lds_b128)
#    define HAVE_ASYNC_LDS 1
#  endif
#endif

static __device__ __forceinline__ void wait_async0() {
#if defined(HAVE_ASYNC_LDS)
#  if __has_builtin(__builtin_amdgcn_s_wait_asynccnt)
    __builtin_amdgcn_s_wait_asynccnt(0);
#  else
    asm volatile("s_wait_asynccnt 0" ::: "memory");
#  endif
#endif
}

// ---------------- weight prep: fp32 [K][Nout] -> bf16 [Np][Kp] (transposed, padded) ----------------
// mode 0: plain. mode 1: head row permutation — padded row kp maps to source row:
//   kp in [0,640)   -> 34+kp   (conv features first)
//   kp in [640,674) -> kp-640  (raw x features)
//   kp >= 674       -> zero
__global__ __launch_bounds__(256) void prep_weight(const float* __restrict__ W,
                                                   bf16_t* __restrict__ O,
                                                   int K, int Nout, int Kp, int Np, int mode) {
    int i = blockIdx.x * 256 + threadIdx.x;
    if (i >= Kp * Np) return;
    int n = i / Kp, kp = i - n * Kp;
    int ksrc;
    if (mode == 1) {
        if (kp < 640)      ksrc = 34 + kp;
        else if (kp < 674) ksrc = kp - 640;
        else               ksrc = -1;
    } else {
        ksrc = (kp < K) ? kp : -1;
    }
    float v = (ksrc >= 0 && n < Nout) ? W[(size_t)ksrc * Nout + n] : 0.0f;
    O[i] = (bf16_t)v;
}

// ---------------- generic WMMA GEMM: C = act(A(MxK) @ W(KxN) + b) ----------------
// A: fp32, row stride ldA (16B-aligned rows), Kp multiple of 64, zero-padded.
// WT: bf16 [Np][Kp] (n-major, transposed). C: fp32, stride ldC; cols [Nout,ldC) zeroed.
// Software-pipelined, LDS double-buffered; B tiles streamed with async->LDS.
__global__ __launch_bounds__(256) void gemm_bias_act(
    const float* __restrict__ A, int ldA,
    const bf16_t* __restrict__ WT, int Kp, int Np,
    const float* __restrict__ bias, int Nout,
    float* __restrict__ C, int ldC, int act) {
    __shared__ bf16_t sA[2][MT * KC];   // 2 x 16 KB
    __shared__ bf16_t sB[2][NT * KC];   // 2 x  8 KB, layout [n][k]
    const int tid  = threadIdx.x;
    const int m0g  = blockIdx.x * MT;
    const int n0g  = blockIdx.y * NT;
    const int wave = tid >> 5;
    const int lane = tid & 31;

    // --- staging helpers (compile-time trip counts; uniform control flow) ---
    float4 ra[8];
    auto loadA = [&](int k0) {                 // 8 independent b128 loads -> regs
        #pragma unroll
        for (int p = 0; p < 8; ++p) {
            int i = p * 256 + tid;
            int r = i >> 4, c4 = i & 15;       // 16 float4 per 64-float row
            ra[p] = *(const float4*)(A + (size_t)(m0g + r) * ldA + k0 + c4 * 4);
        }
    };
    auto storeA = [&](int buf) {               // convert + packed b64 LDS stores
        #pragma unroll
        for (int p = 0; p < 8; ++p) {
            int i = p * 256 + tid;
            int r = i >> 4, c4 = i & 15;
            union { bf16_t h[4]; uint2 u; } pk;
            pk.h[0] = (bf16_t)ra[p].x; pk.h[1] = (bf16_t)ra[p].y;
            pk.h[2] = (bf16_t)ra[p].z; pk.h[3] = (bf16_t)ra[p].w;
            *(uint2*)&sA[buf][r * KC + c4 * 4] = pk.u;
        }
    };
    auto loadB = [&](int k0, int buf) {        // 2 async b128 per thread
        #pragma unroll
        for (int p = 0; p < 2; ++p) {
            int i  = p * 256 + tid;
            int n  = i >> 3, c8 = i & 7;       // 8 uint4 per 64-bf16 row
            int gn = n0g + n;
            if (gn >= Np) gn = Np - 1;         // clamp: padded rows never feed written cols
            const bf16_t* src = WT + (size_t)gn * Kp + k0 + c8 * 8;
#if defined(HAVE_ASYNC_LDS)
            __builtin_amdgcn_global_load_async_to_lds_b128(
                (__attribute__((address_space(1))) v4i*)(uintptr_t)src,
                (__attribute__((address_space(3))) v4i*)(uintptr_t)&sB[buf][n * KC + c8 * 8],
                0, 0);
#else
            *(uint4*)&sB[buf][n * KC + c8 * 8] = *(const uint4*)src;
#endif
        }
    };

    v8f acc[4];
    #pragma unroll
    for (int t = 0; t < 4; ++t) acc[t] = (v8f){0.f,0.f,0.f,0.f,0.f,0.f,0.f,0.f};

    const int nK = Kp / KC;

    // ---- pipeline prologue: fill buffer 0, pre-load A chunk 1 into regs ----
    loadA(0);
    loadB(0, 0);
    storeA(0);
    if (nK > 1) loadA(KC);
    wait_async0();
    __syncthreads();

    for (int kt = 0; kt < nK; ++kt) {
        const int cur = kt & 1;
        const int nxt = cur ^ 1;
        if (kt + 1 < nK) {                     // stage next tile while computing
            loadB((kt + 1) * KC, nxt);
            storeA(nxt);                       // regs hold chunk kt+1
            if (kt + 2 < nK) loadA((kt + 2) * KC);
        }

        const int arow = lane & 15;
        const int koff = (lane >> 4) << 3;     // A frag: lane<16 -> K{0..7,16..23}; else K{8..15,24..31}
        const int bcol = lane & 15;
        const int kb2  = (lane >> 4) << 3;     // B frag: lane<16 -> K{0..7,16..23}; else K{8..15,24..31}
        #pragma unroll
        for (int ks = 0; ks < 2; ++ks) {
            const bf16_t* ap = &sA[cur][(wave * 16 + arow) * KC + ks * 32 + koff];
            v16bf af;
            #pragma unroll
            for (int j = 0; j < 8; ++j) { af[j] = ap[j]; af[8 + j] = ap[16 + j]; }
            #pragma unroll
            for (int t = 0; t < 4; ++t) {
                const bf16_t* bp = &sB[cur][(t * 16 + bcol) * KC + ks * 32 + kb2];
                v16bf bfv;
                #pragma unroll
                for (int j = 0; j < 8; ++j) { bfv[j] = bp[j]; bfv[8 + j] = bp[16 + j]; }
                acc[t] = __builtin_amdgcn_wmma_f32_16x16x32_bf16(
                    false, af, false, bfv, (short)0, acc[t], false, false);
            }
        }
        wait_async0();
        __syncthreads();                       // next buffer ready / current safe to reuse
    }

    // epilogue: C layout per ISA (lane<16: M=r; lane>=16: M=r+8), N = lane&15
    const int ncol = lane & 15;
    const int mhi  = (lane >> 4) << 3;
    #pragma unroll
    for (int t = 0; t < 4; ++t) {
        #pragma unroll
        for (int r = 0; r < 8; ++r) {
            int m = m0g + wave * 16 + mhi + r;
            int gcol = n0g + t * 16 + ncol;
            if (gcol < Nout) {
                float v = acc[t][r] + (bias ? bias[gcol] : 0.0f);
                if (act == 1) v = (v > 0.0f) ? v : (__expf(v) - 1.0f);
                C[(size_t)m * ldC + gcol] = v;
            } else if (gcol < ldC) {
                C[(size_t)m * ldC + gcol] = 0.0f;   // keep padding exactly zero
            }
        }
    }
}

// ---------------- pad x (N x 34) -> (N x 64, zero padded) ----------------
__global__ __launch_bounds__(256) void pad_x_kernel(const float* __restrict__ x,
                                                    float* __restrict__ xp) {
    int i = blockIdx.x * 256 + threadIdx.x;
    if (i >= NQ * 64) return;
    int r = i >> 6, c = i & 63;
    xp[i] = (c < 34) ? x[(size_t)r * 34 + c] : 0.0f;
}

// ---------------- head-feature scaffold: f[:, 640:674]=x, f[:, 674:704]=0 ----------------
// layout: [conv0(128) .. conv4(128) | x(34) | pad(30)]  (conv cols written by conv GEMMs)
__global__ __launch_bounds__(256) void assemble_feat_kernel(const float* __restrict__ x,
                                                            float* __restrict__ f) {
    int i = blockIdx.x * 256 + threadIdx.x;
    if (i >= NQ * 64) return;
    int r = i >> 6, c = i & 63;
    f[(size_t)r * 704 + 640 + c] = (c < 34) ? x[(size_t)r * 34 + c] : 0.0f;
}

// ---------------- kNN (K=8) in learned 4-D space, per graph of 1024 ----------------
__global__ __launch_bounds__(256) void knn_kernel(const float* __restrict__ s,
                                                  int* __restrict__ idx,
                                                  float* __restrict__ wts) {
    __shared__ float ss[GSIZE * 4];   // 16 KB
    const int g = blockIdx.x;
    const float* sg = s + (size_t)g * GSIZE * 4;
    for (int i = threadIdx.x; i < GSIZE * 4; i += 256) ss[i] = sg[i];
    __syncthreads();

    const int i = blockIdx.y * 256 + threadIdx.x;   // row within graph
    const float si0 = ss[i * 4 + 0], si1 = ss[i * 4 + 1];
    const float si2 = ss[i * 4 + 2], si3 = ss[i * 4 + 3];

    float bd[8]; int bi[8];
    #pragma unroll
    for (int q = 0; q < 8; ++q) { bd[q] = 3.4e38f; bi[q] = 0; }

    for (int j = 0; j < GSIZE; ++j) {
        float d0 = si0 - ss[j * 4 + 0];
        float d1 = si1 - ss[j * 4 + 1];
        float d2 = si2 - ss[j * 4 + 2];
        float d3 = si3 - ss[j * 4 + 3];
        float d = d0 * d0 + d1 * d1 + d2 * d2 + d3 * d3;
        if (d < bd[7]) {
            bd[7] = d; bi[7] = j;
            #pragma unroll
            for (int q = 7; q > 0; --q) {
                if (bd[q] < bd[q - 1]) {
                    float td = bd[q]; bd[q] = bd[q - 1]; bd[q - 1] = td;
                    int   ti = bi[q]; bi[q] = bi[q - 1]; bi[q - 1] = ti;
                }
            }
        }
    }
    const size_t base = ((size_t)g * GSIZE + i) * 8;
    #pragma unroll
    for (int q = 0; q < 8; ++q) {
        idx[base + q] = bi[q];
        wts[base + q] = __expf(-10.0f * bd[q]);
    }
}

// ---------------- aggregate messages + build conv GEMM input [x(128) | mean(32) | max(32)] ----------------
__global__ __launch_bounds__(256) void aggregate_kernel(const float* __restrict__ h,
                                                        const int* __restrict__ idx,
                                                        const float* __restrict__ wts,
                                                        const float* __restrict__ xin, int ldX,
                                                        float* __restrict__ feat) {
    int i = blockIdx.x * 256 + threadIdx.x;
    if (i >= NQ) return;
    const size_t gbase = (size_t)(i >> 10) << 10;   // graph start node
    float mean[32], mx[32];
    #pragma unroll
    for (int c = 0; c < 32; ++c) { mean[c] = 0.0f; mx[c] = -3.4e38f; }
    #pragma unroll
    for (int k = 0; k < 8; ++k) {
        int j   = idx[(size_t)i * 8 + k];
        float w = wts[(size_t)i * 8 + k];
        const float* hj = h + (gbase + j) * 32;
        #pragma unroll
        for (int c = 0; c < 32; ++c) {
            float m = hj[c] * w;
            mean[c] += m;
            mx[c] = fmaxf(mx[c], m);
        }
    }
    float* fr = feat + (size_t)i * 192;
    const float* xr = xin + (size_t)i * ldX;
    #pragma unroll
    for (int c = 0; c < 128; ++c) fr[c] = xr[c];
    #pragma unroll
    for (int c = 0; c < 32; ++c) {
        fr[128 + c] = mean[c] * 0.125f;
        fr[160 + c] = mx[c];
    }
}

// ---------------- preds_mom += x[:,1:5] ----------------
__global__ __launch_bounds__(256) void add_mom_kernel(const float* __restrict__ x,
                                                      float* __restrict__ mom) {
    int i = blockIdx.x * 256 + threadIdx.x;
    if (i >= NQ * 4) return;
    int r = i >> 2, c = i & 3;
    mom[i] += x[(size_t)r * 34 + 1 + c];
}

// =====================================================================================
extern "C" void kernel_launch(void* const* d_in, const int* in_sizes, int n_in,
                              void* d_out, int out_size, void* d_ws, size_t ws_size,
                              hipStream_t stream) {
    const float* x      = (const float*)d_in[0];
    const float* n0W1   = (const float*)d_in[2];
    const float* n0b1   = (const float*)d_in[3];
    const float* n0W23  = (const float*)d_in[4];
    const float* n0b23  = (const float*)d_in[5];
    const float* n0W4   = (const float*)d_in[6];
    const float* n0b4   = (const float*)d_in[7];
    const float* cWh    = (const float*)d_in[8];
    const float* cbh    = (const float*)d_in[9];
    const float* cWs    = (const float*)d_in[10];
    const float* cbs    = (const float*)d_in[11];
    const float* cWo    = (const float*)d_in[12];
    const float* cbo    = (const float*)d_in[13];
    const float* HW1    = (const float*)d_in[14];
    const float* Hb1    = (const float*)d_in[15];
    const float* HW23   = (const float*)d_in[16];
    const float* Hb23   = (const float*)d_in[17];
    const float* Wid    = (const float*)d_in[18];
    const float* bid    = (const float*)d_in[19];
    const float* Wreg   = (const float*)d_in[20];
    const float* breg   = (const float*)d_in[21];
    const float* Wch    = (const float*)d_in[22];
    const float* bch    = (const float*)d_in[23];
    float* out = (float*)d_out;

    // ---- workspace bump allocator ----
    char* ws = (char*)d_ws;
    size_t cur = 0;
    auto alloc = [&](size_t bytes) -> char* {
        cur = (cur + 255) & ~(size_t)255;
        char* p = ws + cur;
        cur += bytes;
        return p;
    };
    float* xpad = (float*)alloc((size_t)NQ * 64 * 4);
    float* emb  = (float*)alloc((size_t)NQ * 128 * 4);
    float* tbuf = (float*)alloc((size_t)NQ * 128 * 4);
    float* ubuf = (float*)alloc((size_t)NQ * 128 * 4);
    float* fbuf = (float*)alloc((size_t)NQ * 704 * 4);
    float* hbuf = (float*)alloc((size_t)NQ * 32 * 4);
    float* sbuf = (float*)alloc((size_t)NQ * 4 * 4);
    int*   idxb = (int*)  alloc((size_t)NQ * 8 * 4);
    float* wtsb = (float*)alloc((size_t)NQ * 8 * 4);
    float* feat = (float*)alloc((size_t)NQ * 192 * 4);

    bf16_t* wn0W1  = (bf16_t*)alloc(64 * 128 * 2);
    bf16_t* wn0W23[2]; for (int i = 0; i < 2; ++i) wn0W23[i] = (bf16_t*)alloc(128 * 128 * 2);
    bf16_t* wn0W4  = (bf16_t*)alloc(128 * 128 * 2);
    bf16_t* wcWh[10]; for (int i = 0; i < 10; ++i) wcWh[i] = (bf16_t*)alloc(128 * 32 * 2);
    bf16_t* wcWs[10]; for (int i = 0; i < 10; ++i) wcWs[i] = (bf16_t*)alloc(128 * 16 * 2);
    bf16_t* wcWo[10]; for (int i = 0; i < 10; ++i) wcWo[i] = (bf16_t*)alloc(192 * 128 * 2);
    bf16_t* wHW1[3];  for (int i = 0; i < 3; ++i)  wHW1[i]  = (bf16_t*)alloc(704 * 128 * 2);
    bf16_t* wHW23[6]; for (int i = 0; i < 6; ++i)  wHW23[i] = (bf16_t*)alloc(128 * 128 * 2);
    bf16_t* wWid  = (bf16_t*)alloc(128 * 16 * 2);
    bf16_t* wWreg = (bf16_t*)alloc(128 * 16 * 2);
    bf16_t* wWch  = (bf16_t*)alloc(128 * 16 * 2);

    // ---- host-side helpers (enqueue only) ----
    auto prep = [&](const float* W, bf16_t* O, int K, int Nout, int Kp, int Np, int mode) {
        int tot = Kp * Np;
        prep_weight<<<(tot + 255) / 256, 256, 0, stream>>>(W, O, K, Nout, Kp, Np, mode);
    };
    auto gemm = [&](const float* A, int ldA, const bf16_t* WT, int Kp, int Np,
                    const float* bias, int Nout, float* C, int ldC, int act) {
        dim3 grid(NQ / MT, (Np + NT - 1) / NT);
        gemm_bias_act<<<grid, 256, 0, stream>>>(A, ldA, WT, Kp, Np, bias, Nout, C, ldC, act);
    };

    // ---- convert all weights to transposed/padded bf16 ----
    prep(n0W1, wn0W1, 34, 126, 64, 128, 0);
    for (int j = 0; j < 2; ++j) prep(n0W23 + (size_t)j * 126 * 126, wn0W23[j], 126, 126, 128, 128, 0);
    prep(n0W4, wn0W4, 126, 128, 128, 128, 0);
    for (int i = 0; i < 10; ++i) {
        prep(cWh + (size_t)i * 128 * 32,  wcWh[i], 128, 32,  128, 32, 0);
        prep(cWs + (size_t)i * 128 * 4,   wcWs[i], 128, 4,   128, 16, 0);
        prep(cWo + (size_t)i * 192 * 128, wcWo[i], 192, 128, 192, 128, 0);
    }
    for (int i = 0; i < 3; ++i) prep(HW1 + (size_t)i * 674 * 126, wHW1[i], 674, 126, 704, 128, 1);
    for (int i = 0; i < 6; ++i) prep(HW23 + (size_t)i * 126 * 126, wHW23[i], 126, 126, 128, 128, 0);
    prep(Wid,  wWid,  126, 8, 128, 16, 0);
    prep(Wreg, wWreg, 126, 4, 128, 16, 0);
    prep(Wch,  wWch,  126, 1, 128, 16, 0);

    // ---- embedding MLP4: x -> emb (N x 128) ----
    pad_x_kernel<<<(NQ * 64 + 255) / 256, 256, 0, stream>>>(x, xpad);
    gemm(xpad, 64,  wn0W1,     64, 128, n0b1,         126, tbuf, 128, 1);
    gemm(tbuf, 128, wn0W23[0], 128, 128, n0b23,       126, ubuf, 128, 1);
    gemm(ubuf, 128, wn0W23[1], 128, 128, n0b23 + 126, 126, tbuf, 128, 1);
    gemm(tbuf, 128, wn0W4,     128, 128, n0b4,        128, emb,  128, 0);

    // ---- per-stack convs + heads; conv-l output lands at fbuf column l*128 ----
    for (int st = 0; st < 2; ++st) {
        assemble_feat_kernel<<<(NQ * 64 + 255) / 256, 256, 0, stream>>>(x, fbuf);
        const float* cin = emb;
        int ldin = 128;
        for (int l = 0; l < 5; ++l) {
            int li = st * 5 + l;
            gemm(cin, ldin, wcWh[li], 128, 32, cbh + (size_t)li * 32, 32, hbuf, 32, 0);
            gemm(cin, ldin, wcWs[li], 128, 16, cbs + (size_t)li * 4,  4,  sbuf, 4,  0);
            knn_kernel<<<dim3(NGRAPH, GSIZE / 256), 256, 0, stream>>>(sbuf, idxb, wtsb);
            aggregate_kernel<<<NQ / 256, 256, 0, stream>>>(hbuf, idxb, wtsb, cin, ldin, feat);
            float* outp = fbuf + (size_t)l * 128;   // 16B-aligned column block
            gemm(feat, 192, wcWo[li], 192, 128, cbo + (size_t)li * 128, 128, outp, 704, 0);
            cin = outp;
            ldin = 704;
        }
        auto head = [&](int hi, const bf16_t* WfT, const float* bf_, int NoutF,
                        float* Cout, int ldCout) {
            gemm(fbuf, 704, wHW1[hi], 704, 128, Hb1 + (size_t)hi * 126, 126, tbuf, 128, 1);
            gemm(tbuf, 128, wHW23[hi * 2 + 0], 128, 128, Hb23 + (size_t)(hi * 2 + 0) * 126, 126, ubuf, 128, 1);
            gemm(ubuf, 128, wHW23[hi * 2 + 1], 128, 128, Hb23 + (size_t)(hi * 2 + 1) * 126, 126, tbuf, 128, 1);
            gemm(tbuf, 128, WfT, 128, 16, bf_, NoutF, Cout, ldCout, 0);
        };
        if (st == 0) {
            head(0, wWid, bid, 8, out, 8);                       // preds_id: N x 8
        } else {
            head(1, wWreg, breg, 4, out + (size_t)NQ * 8, 4);    // preds_mom: N x 4
            add_mom_kernel<<<(NQ * 4 + 255) / 256, 256, 0, stream>>>(x, out + (size_t)NQ * 8);
            head(2, wWch, bch, 1, out + (size_t)NQ * 12, 1);     // pred_charge: N x 1
        }
    }
}
